// PixelContrastLoss_71811853189418
// MI455X (gfx1250) — matrix operand order repaired and checked by reading, not compiled, not run on hardware
//
#include <hip/hip_runtime.h>
#include <hip/hip_bf16.h>

// ---------------- problem constants ----------------
#define BB   4
#define DD   256
#define HH   128
#define WW2  128
#define HW   (HH * WW2)         // 16384
#define CC   21
#define AA   (BB * CC)          // 84
#define VV   100
#define NN   (AA * VV)          // 8400
#define GSZ  (NN / CC)          // 400 rows per label group (sorted layout)
#define GTIL (GSZ / 16)         // 25 j-tiles per group
#define NTIL (NN / 16)          // 525
#define PH   64                 // predict H/W
#define INV_TEMP 10.0f
#define SHIFT 10.0f             // logits <= ~10 (unit-norm rows); LSE is shift-invariant
#define NPOS 399.0f             // 4*100 - 1 positives per row (constant by construction)

typedef _Float16 v16h __attribute__((ext_vector_type(16)));
typedef _Float16 h8   __attribute__((ext_vector_type(8)));
typedef float    v8f  __attribute__((ext_vector_type(8)));

union HalfU { v16h v; h8 p[2]; };

// ---------------- workspace layout (bytes) ----------------
#define OFF_FEATST 0                           // [B][HW][D] f16   33,554,432
#define OFF_CFH    ((size_t)33554432)          // [N][D] f16 label-sorted 4,300,800
#define OFF_PRED   (OFF_CFH + 4300800)         // [B*HW] int         262,144
#define OFF_IDX    (OFF_PRED + 262144)         // [A][V] int          33,664
#define OFF_NS     (OFF_IDX + 33664)           // [N] f32 neg_sum     33,664
#define OFF_POSS   (OFF_NS + 33664)            // [N] f32 pos terms   33,664

__device__ __forceinline__ int imin(int a, int b) { return a < b ? a : b; }

__device__ __forceinline__ float rsum16(float v) {
#pragma unroll
  for (int m = 8; m >= 1; m >>= 1) v += __shfl_xor(v, m, 32);
  return v;
}

// ---------- 1) L2-normalize over channels, transpose to [B][HW][D] f16 ----------
__global__ void k_norm_transpose(const float* __restrict__ feats, _Float16* __restrict__ ftT) {
  int gid = blockIdx.x * blockDim.x + threadIdx.x;   // one thread per pixel
  if (gid >= BB * HW) return;
  int b = gid / HW, p = gid % HW;
  const float* src = feats + (size_t)b * DD * HW + p;
  float ss = 0.f;
#pragma unroll 8
  for (int d = 0; d < DD; ++d) { float x = src[(size_t)d * HW]; ss += x * x; }
  float inv = 1.0f / fmaxf(sqrtf(ss), 1e-12f);
  _Float16* dst = ftT + (size_t)gid * DD;
  for (int d0 = 0; d0 < DD; d0 += 8) {
    h8 v;
#pragma unroll
    for (int k = 0; k < 8; ++k) v[k] = (_Float16)(src[(size_t)(d0 + k) * HW] * inv);
    *(h8*)(dst + d0) = v;                            // 16B store per 8 channels
  }
}

// ---------- 2) bilinear upsample (64->128, half-pixel centers, edge clamp) + argmax ----------
__global__ void k_pred(const float* __restrict__ predict, int* __restrict__ pred) {
  int gid = blockIdx.x * blockDim.x + threadIdx.x;
  if (gid >= BB * HW) return;
  int b = gid / HW, p = gid % HW, y = p / WW2, x = p % WW2;
  float fy = fminf(fmaxf(0.5f * (float)y - 0.25f, 0.f), (float)(PH - 1));
  float fx = fminf(fmaxf(0.5f * (float)x - 0.25f, 0.f), (float)(PH - 1));
  int y0 = (int)fy, x0 = (int)fx;
  int y1 = imin(y0 + 1, PH - 1), x1 = imin(x0 + 1, PH - 1);
  float wy = fy - (float)y0, wx = fx - (float)x0;
  float best = -3.0e38f; int bc = 0;
  const float* pb = predict + (size_t)b * CC * PH * PH;
  for (int c = 0; c < CC; ++c) {
    const float* pc = pb + (size_t)c * PH * PH;
    float v00 = pc[y0 * PH + x0], v01 = pc[y0 * PH + x1];
    float v10 = pc[y1 * PH + x0], v11 = pc[y1 * PH + x1];
    float v = (1.f - wy) * ((1.f - wx) * v00 + wx * v01) + wy * ((1.f - wx) * v10 + wx * v11);
    if (v > best) { best = v; bc = c; }
  }
  pred[gid] = bc;
}

// ---------- 3) deterministic index selection per (image,class) ----------
__global__ void k_select(const int* __restrict__ pred, const int* __restrict__ labels,
                         int* __restrict__ idx) {
  __shared__ int sh_h[257], sh_e[257];
  int m = blockIdx.x;                // 0..83
  int b = m / CC, c = m % CC;
  int t = threadIdx.x;               // 256 threads, 64 pixels each
  const int* pd = pred + (size_t)b * HW;
  const int* lb = labels + (size_t)b * HW;
  int p0 = t * 64;
  int hc = 0, ec = 0;
  for (int k = 0; k < 64; ++k) {
    int p = p0 + k;
    bool hit = (pd[p] == c);
    bool eq  = (lb[p] == c);
    hc += (hit && !eq); ec += (hit && eq);
  }
  sh_h[t] = hc; sh_e[t] = ec;
  __syncthreads();
  if (t == 0) {
    int rh = 0, re = 0;
    for (int i = 0; i < 256; ++i) {
      int h = sh_h[i], e = sh_e[i];
      sh_h[i] = rh; sh_e[i] = re;
      rh += h; re += e;
    }
    sh_h[256] = rh; sh_e[256] = re;
  }
  __syncthreads();
  int h_bef = sh_h[t], e_bef = sh_e[t];
  int nh = sh_h[256], ne = sh_e[256];
  int half = VV / 2, hk;
  if (nh >= half && ne >= half) hk = half;
  else if (nh >= half)          hk = VV - ne;
  else                          hk = nh;
  int ek = VV - hk;
  int total_sel = imin(nh, hk) + imin(ne, ek);
  for (int k = 0; k < 64; ++k) {
    int p = p0 + k;
    bool hit = (pd[p] == c);
    bool eq  = (lb[p] == c);
    int  sel_before = imin(h_bef, hk) + imin(e_bef, ek);
    bool sel = false;
    if (hit && !eq) { sel = (h_bef < hk); h_bef++; }
    else if (hit)   { sel = (e_bef < ek); e_bef++; }
    int pos = sel ? sel_before : (total_sel + (p - sel_before));
    if (pos < VV) idx[m * VV + pos] = p;
  }
}

// ---------- 4) gather cf rows (f16) into LABEL-SORTED layout ----------
// permuted row q: group g=q/400 (label), t=q%400, original n = g + 21*t
// (valid since label(n) = (n%84)%21 = n%21 because 84 = 4*21)
__global__ void k_gather(const _Float16* __restrict__ ftT, const int* __restrict__ idx,
                         _Float16* __restrict__ cfh) {
  int gid = blockIdx.x * blockDim.x + threadIdx.x;
  int q = gid >> 5, lane = gid & 31;
  if (q >= NN) return;
  int g = q / GSZ, t = q % GSZ;
  int n = g + CC * t;
  int a = n % AA, v = n / AA, b = a / CC;
  int p = idx[a * VV + v];
  const float4* src = (const float4*)(ftT + ((size_t)(b * HW + p)) * DD) + lane;
  float4* dst = (float4*)(cfh + (size_t)q * DD) + lane;
  *dst = *src;                       // 16B per lane = 512B per row
}

// ---------- WMMA fragment helpers ----------
__device__ __forceinline__ void load_afrags(const _Float16* cfh, int i0, int Lm, int hi,
                                            v16h afr[8]) {
  const _Float16* arow = cfh + (size_t)(i0 + Lm) * DD;
#pragma unroll
  for (int kc = 0; kc < 8; ++kc) {
    HalfU u;
    u.p[0] = *(const h8*)(arow + kc * 32 + 16 * hi);
    u.p[1] = *(const h8*)(arow + kc * 32 + 16 * hi + 8);
    afr[kc] = u.v;
  }
}

__device__ __forceinline__ v8f tile_gemm(const v16h afr[8], const _Float16* cfh,
                                         int j0, int Lm, int hi) {
  const _Float16* brow = cfh + (size_t)(j0 + Lm) * DD;
  v16h bfr[8];
#pragma unroll
  for (int kc = 0; kc < 8; ++kc)     // preload: one s_clause of 8x b256 worth of loads
    bfr[kc] = *(const v16h*)(brow + kc * 32 + 16 * hi);
  v8f acc = {};
#pragma unroll
  for (int kc = 0; kc < 8; ++kc)     // back-to-back WMMAs
    acc = __builtin_amdgcn_wmma_f32_16x16x32_f16(false, afr[kc], false, bfr[kc],
                                                 (short)0, acc, false, false);
  return acc;
}

// ---------- 5) pass 1: full-row exp-sum and own-group exp-sum (fixed shift) ----------
// neg_sum = full_sum - group_sum (group block = same label, includes diagonal)
// one wave owns one i-tile; 8 waves of a block sweep identical j sequence (B hits L0)
__global__ void k_pass1(const _Float16* __restrict__ cfh, float* __restrict__ row_ns) {
  int w = threadIdx.x >> 5, lane = threadIdx.x & 31;
  int it = blockIdx.x * 8 + w;
  if (it >= NTIL) return;
  int i0 = it * 16;
  int Lm = lane & 15, hi = lane >> 4;
  int myg = i0 / GSZ;                 // i-tile never crosses a group (16 | 400)

  v16h afr[8];
  load_afrags(cfh, i0, Lm, hi, afr);

  float s_all[8], s_grp[8];
#pragma unroll
  for (int r = 0; r < 8; ++r) { s_all[r] = 0.f; s_grp[r] = 0.f; }

  for (int jt = 0; jt < NTIL; ++jt) {
    int j0 = jt * 16;
    v8f acc = tile_gemm(afr, cfh, j0, Lm, hi);
    bool same = (j0 / GSZ) == myg;    // uniform across wave
#pragma unroll
    for (int r = 0; r < 8; ++r) {
      float e = __expf(acc[r] * INV_TEMP - SHIFT);
      s_all[r] += e;
      if (same) s_grp[r] += e;
    }
  }
#pragma unroll
  for (int r = 0; r < 8; ++r) {
    float a = rsum16(s_all[r]);
    float g = rsum16(s_grp[r]);
    if (Lm == 0) row_ns[i0 + r + 8 * hi] = a - g;   // lanes 0 and 16 -> rows r, r+8
  }
}

// ---------- 6) pass 2: positive log-prob terms (only own group's 25 j-tiles) ----------
__global__ void k_pass2(const _Float16* __restrict__ cfh, const float* __restrict__ row_ns,
                        float* __restrict__ pos_sum) {
  int w = threadIdx.x >> 5, lane = threadIdx.x & 31;
  int it = blockIdx.x * 8 + w;
  if (it >= NTIL) return;
  int i0 = it * 16;
  int Lm = lane & 15, hi = lane >> 4;
  int myg = i0 / GSZ;

  v16h afr[8];
  load_afrags(cfh, i0, Lm, hi, afr);

  int gi[8];
  float ns[8], ps[8];
#pragma unroll
  for (int r = 0; r < 8; ++r) {
    gi[r] = i0 + r + 8 * hi;
    ns[r] = row_ns[gi[r]];
    ps[r] = 0.f;
  }

  for (int jj = 0; jj < GTIL; ++jj) {          // only same-label block: all pos but diag
    int j0 = myg * GSZ + jj * 16;
    v8f acc = tile_gemm(afr, cfh, j0, Lm, hi);
    int gj = j0 + Lm;
#pragma unroll
    for (int r = 0; r < 8; ++r) {
      float lp = acc[r] * INV_TEMP - SHIFT;
      float term = (gj != gi[r]) ? (lp - __logf(__expf(lp) + ns[r])) : 0.f;
      ps[r] += term;
    }
  }
#pragma unroll
  for (int r = 0; r < 8; ++r) {
    float s = rsum16(ps[r]);
    if (Lm == 0) pos_sum[i0 + r + 8 * hi] = s;
  }
}

// ---------- 7) final reduction: loss = -mean_i(pos_sum_i / 399) ----------
__global__ void k_finalize(const float* __restrict__ pos_sum, float* __restrict__ out) {
  __shared__ float sh[256];
  int t = threadIdx.x;
  float s = 0.f;
  for (int i = t; i < NN; i += 256) s += pos_sum[i];
  sh[t] = s;
  __syncthreads();
  for (int off = 128; off > 0; off >>= 1) {
    if (t < off) sh[t] += sh[t + off];
    __syncthreads();
  }
  if (t == 0) out[0] = -(sh[0] / NPOS) / (float)NN;
}

extern "C" void kernel_launch(void* const* d_in, const int* in_sizes, int n_in,
                              void* d_out, int out_size, void* d_ws, size_t ws_size,
                              hipStream_t stream) {
  (void)in_sizes; (void)n_in; (void)out_size; (void)ws_size;
  const float* feats   = (const float*)d_in[0];
  const float* predict = (const float*)d_in[1];
  const int*   labels  = (const int*)d_in[2];
  float* out = (float*)d_out;

  char* ws = (char*)d_ws;
  _Float16* ftT   = (_Float16*)(ws + OFF_FEATST);
  _Float16* cfh   = (_Float16*)(ws + OFF_CFH);
  int*      pred  = (int*)(ws + OFF_PRED);
  int*      idx   = (int*)(ws + OFF_IDX);
  float*    rns   = (float*)(ws + OFF_NS);
  float*    poss  = (float*)(ws + OFF_POSS);

  const int gemmBlocks = (NTIL + 7) / 8;       // 66 blocks of 8 waves

  k_norm_transpose<<<(BB * HW) / 256, 256, 0, stream>>>(feats, ftT);
  k_pred<<<(BB * HW) / 256, 256, 0, stream>>>(predict, pred);
  k_select<<<AA, 256, 0, stream>>>(pred, labels, idx);
  k_gather<<<(NN * 32) / 256, 256, 0, stream>>>(ftT, idx, cfh);
  k_pass1<<<gemmBlocks, 256, 0, stream>>>(cfh, rns);
  k_pass2<<<gemmBlocks, 256, 0, stream>>>(cfh, rns, poss);
  k_finalize<<<1, 256, 0, stream>>>(poss, out);
}
